// SGENet4_79731772883222
// MI455X (gfx1250) — matrix operand order repaired and checked by reading, not compiled, not run on hardware
//
#include <hip/hip_runtime.h>
#include <hip/hip_bf16.h>

typedef __attribute__((ext_vector_type(2)))  float  v2f;
typedef __attribute__((ext_vector_type(8)))  float  v8f;
typedef __attribute__((ext_vector_type(16))) __bf16 v16bf;

#define BATCH 32
#define NPTS  2048
#define CH    128
#define KNN   6
#define NTGT  16
#define NEDGE 96   // NTGT*KNN

union BF16Frag { v16bf v; uint4 u[2]; };

// ---------------------------------------------------------------------------
// A-fragment (16x32 bf16) built from an fp32 LDS tile, row stride CH floats.
// Layout (ISA 7.12.2): lane l holds row M=l%16; lanes 0-15 get K = kBase+[0..7]
// and kBase+[16..23]; lanes 16-31 get kBase+[8..15] and kBase+[24..31].
__device__ __forceinline__ v16bf ldsAfragF32(const float* base, int row,
                                             int kBase, int half) {
  const float* p = base + row * CH + kBase + (half << 3);
  float4 a0 = *(const float4*)(p);
  float4 a1 = *(const float4*)(p + 4);
  float4 b0 = *(const float4*)(p + 16);
  float4 b1 = *(const float4*)(p + 20);
  v16bf v;
  v[0]  = (__bf16)a0.x; v[1]  = (__bf16)a0.y; v[2]  = (__bf16)a0.z; v[3]  = (__bf16)a0.w;
  v[4]  = (__bf16)a1.x; v[5]  = (__bf16)a1.y; v[6]  = (__bf16)a1.z; v[7]  = (__bf16)a1.w;
  v[8]  = (__bf16)b0.x; v[9]  = (__bf16)b0.y; v[10] = (__bf16)b0.z; v[11] = (__bf16)b0.w;
  v[12] = (__bf16)b1.x; v[13] = (__bf16)b1.y; v[14] = (__bf16)b1.z; v[15] = (__bf16)b1.w;
  return v;
}

// A-fragment (16x32 bf16) from a bf16 LDS tile (used for h in lin2).
__device__ __forceinline__ v16bf ldsAfragBF16(const __bf16* base, int row,
                                              int kBase, int half) {
  const __bf16* p = base + row * CH + kBase + (half << 3);
  BF16Frag f;
  f.u[0] = *(const uint4*)(p);
  f.u[1] = *(const uint4*)(p + 16);
  return f.v;
}

// B-fragment (32x16 bf16) from a row-major weight matrix W[n][k] (ldw=fan_in),
// B[k][n] = W[n][k]. Lane l holds column N=l%16; lanes 0-15 hold
// K = kBase+[0..7],[16..23], lanes 16-31 hold kBase+[8..15],[24..31].
__device__ __forceinline__ v16bf gblBfragBF16(const __bf16* w, int ldw,
                                              int nBase, int kBase, int lane) {
  const __bf16* p = w + (size_t)(nBase + (lane & 15)) * ldw + kBase + ((lane >> 4) << 3);
  BF16Frag f;
  f.u[0] = *(const uint4*)(p);
  f.u[1] = *(const uint4*)(p + 16);
  return f.v;
}

__device__ __forceinline__ v8f wmma_bf16(v16bf a, v16bf b, v8f c) {
  return __builtin_amdgcn_wmma_f32_16x16x32_bf16(false, a, false, b,
                                                 (short)0, c, false, false);
}

// LDS byte offset of a generic pointer into shared memory: the LDS flat
// aperture maps addr[31:0] directly to the LDS offset.
__device__ __forceinline__ unsigned lds_off(const void* p) {
  return (unsigned)(size_t)p;
}

// Async global->LDS copy of 16 bytes per lane (GVS mode), ASYNCcnt-tracked.
__device__ __forceinline__ void async_copy_b128(unsigned ldsDst, unsigned gOff,
                                                const void* base) {
  asm volatile("global_load_async_to_lds_b128 %0, %1, %2"
               :: "v"(ldsDst), "v"(gOff), "s"(base) : "memory");
}
__device__ __forceinline__ void async_wait0() {
  asm volatile("s_wait_asynccnt 0x0" ::: "memory");
}

// ---------------------------------------------------------------------------
// K0a: per-point squared norm (one wave per point)
__global__ void sq_kernel(const float* __restrict__ x, float* __restrict__ sq) {
  int row  = blockIdx.x * 8 + (threadIdx.x >> 5);
  int lane = threadIdx.x & 31;
  const float4 v = *(const float4*)(x + (size_t)row * CH + lane * 4);
  float s = v.x * v.x + v.y * v.y + v.z * v.z + v.w * v.w;
#pragma unroll
  for (int off = 16; off > 0; off >>= 1) s += __shfl_xor(s, off, 32);
  if (lane == 0) sq[row] = s;
}

// K0b: fp32 -> bf16 weight conversion
__global__ void cvt_kernel(const float* __restrict__ src, __bf16* __restrict__ dst, int n) {
  int i = blockIdx.x * blockDim.x + threadIdx.x;
  if (i < n) dst[i] = (__bf16)src[i];
}

// ---------------------------------------------------------------------------
// K1: kNN via fp32 WMMA Gram tiles + per-row top-6. One wave per 16 rows.
// B-fragments are double-buffered so loads of tile t+1 overlap the WMMA chain
// and top-k scan of tile t.
__global__ void __launch_bounds__(32) knn_kernel(const float* __restrict__ x,
                                                 const float* __restrict__ sq,
                                                 int* __restrict__ idx) {
  __shared__ float tile[16 * 16];
  const int b    = blockIdx.x >> 7;     // / (NPTS/16)
  const int rt   = blockIdx.x & 127;
  const int lane = threadIdx.x;
  const int half = lane >> 4;
  const int l16  = lane & 15;
  const int rowG = rt * 16 + l16;
  const float* xb  = x  + (size_t)b * NPTS * CH;
  const float* sqb = sq + (size_t)b * NPTS;

  // A fragments for this row tile (fp32 16x4 layout: lane holds M=l%16,
  // VGPR v -> K = 4c + 2*half + v), held in VGPRs for the whole kernel.
  v2f afr[32];
  {
    const float* xr = xb + (size_t)rowG * CH + (half << 1);
#pragma unroll
    for (int c = 0; c < 32; ++c) afr[c] = *(const v2f*)(xr + c * 4);
  }

  float best[KNN]; int bid[KNN];
#pragma unroll
  for (int t = 0; t < KNN; ++t) { best[t] = 3.0e38f; bid[t] = 0; }

  auto loadB = [&](int ct, v2f* buf) {
    const float* xc = xb + (size_t)(ct * 16 + l16) * CH + (half << 1);
#pragma unroll
    for (int c = 0; c < 32; ++c) buf[c] = *(const v2f*)(xc + c * 4);
  };

  auto procTile = [&](int ct, const v2f* buf) {
    v8f acc = {};
#pragma unroll
    for (int c = 0; c < 32; ++c)
      acc = __builtin_amdgcn_wmma_f32_16x16x4_f32(false, afr[c], false, buf[c],
                                                  (short)0, acc, false, false);
    // C/D layout: lane holds N=l%16, rows M = 8*half + v
#pragma unroll
    for (int v = 0; v < 8; ++v) tile[(half * 8 + v) * 16 + l16] = acc[v];
    __syncthreads();
    if (lane < 16) {
      const int cb = ct * 16;
#pragma unroll
      for (int jj = 0; jj < 16; ++jj) {
        int col = cb + jj;
        float d = sqb[col] - 2.0f * tile[lane * 16 + jj];  // sq_i constant per row
        int mi = 0; float mv = best[0];
#pragma unroll
        for (int t = 1; t < KNN; ++t) { if (best[t] > mv) { mv = best[t]; mi = t; } }
        if ((col != rowG) && (d < mv)) { best[mi] = d; bid[mi] = col; }
      }
    }
    __syncthreads();
  };

  v2f bufA[32], bufB[32];
  loadB(0, bufA);
  for (int ct = 0; ct < NPTS / 16; ct += 2) {
    loadB(ct + 1, bufB);                       // prefetch odd tile
    procTile(ct, bufA);
    if (ct + 2 < NPTS / 16) loadB(ct + 2, bufA);  // prefetch next even tile
    procTile(ct + 1, bufB);
  }

  if (lane < 16) {
    int* op = idx + ((size_t)b * NPTS + rowG) * KNN;
#pragma unroll
    for (int t = 0; t < KNN; ++t) op[t] = bid[t];
  }
}

// ---------------------------------------------------------------------------
// K2: per-edge Q/K attention + FullEdgeConv message + mean aggregation.
// 192 threads (6 waves); block owns 16 targets -> 96 edges; each wave owns a
// 16-edge M-tile. All GEMMs are bf16 WMMA with f32 accumulation.
__global__ void __launch_bounds__(192) edge_kernel(
    const float* __restrict__ x, const int* __restrict__ idx,
    const __bf16* __restrict__ W1h, const float* __restrict__ b1,
    const __bf16* __restrict__ W2h, const float* __restrict__ b2,
    const __bf16* __restrict__ Wqh, const float* __restrict__ bq,
    const __bf16* __restrict__ Wkh, const float* __restrict__ bk,
    float* __restrict__ out) {
  extern __shared__ char smem[];
  float*  lXi = (float*)smem;                           // 16*128 f32   (8 KB)
  float*  lXj = (float*)(smem + 8192);                  // 96*128 f32   (48 KB)
  float*  lQM = (float*)(smem + 8192 + 49152);          // 96*128 f32: Q, then m (48 KB)
  float*  lKf = (float*)(smem + 8192 + 2 * 49152);      // 96*128 f32: K         (48 KB)
  __bf16* lHf = (__bf16*)(smem + 8192 + 2 * 49152);     // 96*128 bf16: h (reuses K region)
  float*  lEW = (float*)(smem + 8192 + 3 * 49152);      // 96*4 f32 edge weights (1.5 KB)

  const int tid  = threadIdx.x;
  const int w    = tid >> 5;
  const int lane = tid & 31;
  const int half = lane >> 4;
  const int l16  = lane & 15;
  const int b    = blockIdx.x >> 7;
  const int t0   = (blockIdx.x & 127) * NTGT;
  const float* xb = x + (size_t)b * NPTS * CH;

  // ---- async gather: x_i (deduped targets) and x_j (neighbors) into LDS ----
  for (int q = tid; q < NTGT * 32; q += 192) {
    int t = q >> 5, c4 = (q & 31) << 2;
    unsigned dst = lds_off(lXi + t * CH + c4);
    unsigned src = (unsigned)(((t0 + t) * CH + c4) * 4);
    async_copy_b128(dst, src, xb);
  }
  for (int q = tid; q < NEDGE * 32; q += 192) {
    int e = q >> 5, c4 = (q & 31) << 2;
    int t = e / KNN, s = e - t * KNN;
    int j = idx[((size_t)b * NPTS + t0 + t) * KNN + s];
    unsigned dst = lds_off(lXj + e * CH + c4);
    unsigned src = (unsigned)((j * CH + c4) * 4);
    async_copy_b128(dst, src, xb);
  }

  // Warm the bf16 weight block (Wq|Wk|W1|W2 contiguous, 224 KB) while the
  // async tensor-data copies are in flight.
  for (int q = tid; q < 1792; q += 192)
    __builtin_prefetch((const char*)Wqh + (size_t)q * 128, 0, 0);

  async_wait0();
  __syncthreads();

  const int eBase = w * 16;
  const int eRow  = eBase + l16;          // A row for x_j / h
  const int tRow  = (eBase + l16) / KNN;  // A row for deduped x_i

  // A fragments for qk_in/msg_in, built once (invariant across all N-tiles).
  v16bf ajf[4], aif[4];
#pragma unroll
  for (int kc = 0; kc < 4; ++kc) {
    ajf[kc] = ldsAfragF32(lXj, eRow, kc * 32, half);
    aif[kc] = ldsAfragF32(lXi, tRow, kc * 32, half);
  }

  // ---- Q and K: qk_in = [x_j, x_i] => W[:, :128]*x_j + W[:, 128:]*x_i ----
  for (int nt = 0; nt < 8; ++nt) {
    v16bf wf[8];
#pragma unroll
    for (int kc = 0; kc < 4; ++kc) {
      wf[kc]     = gblBfragBF16(Wqh, 256, nt * 16, kc * 32, lane);
      wf[kc + 4] = gblBfragBF16(Wqh, 256, nt * 16, 128 + kc * 32, lane);
    }
    v8f accQ = {};
#pragma unroll
    for (int kc = 0; kc < 4; ++kc) {
      accQ = wmma_bf16(ajf[kc], wf[kc], accQ);
      accQ = wmma_bf16(aif[kc], wf[kc + 4], accQ);
    }
#pragma unroll
    for (int kc = 0; kc < 4; ++kc) {
      wf[kc]     = gblBfragBF16(Wkh, 256, nt * 16, kc * 32, lane);
      wf[kc + 4] = gblBfragBF16(Wkh, 256, nt * 16, 128 + kc * 32, lane);
    }
    v8f accK = {};
#pragma unroll
    for (int kc = 0; kc < 4; ++kc) {
      accK = wmma_bf16(ajf[kc], wf[kc], accK);
      accK = wmma_bf16(aif[kc], wf[kc + 4], accK);
    }
    float bqv = bq[nt * 16 + l16];
    float bkv = bk[nt * 16 + l16];
#pragma unroll
    for (int v = 0; v < 8; ++v) {
      int e = eBase + half * 8 + v;
      lQM[e * CH + nt * 16 + l16] = accQ[v] + bqv;
      lKf[e * CH + nt * 16 + l16] = accK[v] + bkv;
    }
  }

  // ---- edge weights: ew[h][g] = (Q_h . K_g)/sqrt(O). Lane: e=l%16, h=half.
  {
    int e = eBase + l16;
    const float* qr = lQM + e * CH + half * 64;
    const float* k0 = lKf + e * CH;
    const float* k1 = lKf + e * CH + 64;
    float a0 = 0.f, a1 = 0.f;
#pragma unroll
    for (int d = 0; d < 64; ++d) {
      float qd = qr[d];
      a0 += qd * k0[d];
      a1 += qd * k1[d];
    }
    const float sc = 0.088388347648318447f;  // 1/sqrt(128)
    lEW[e * 4 + half * 2 + 0] = a0 * sc;
    lEW[e * 4 + half * 2 + 1] = a1 * sc;
  }
  __syncthreads();  // lHf below overlays other waves' lKf rows

  // ---- h = relu(msg_in @ W1^T + b1) + x_i, msg_in = [x_i, x_j] ----
  for (int nt = 0; nt < 8; ++nt) {
    v16bf wf[8];
#pragma unroll
    for (int kc = 0; kc < 4; ++kc) {
      wf[kc]     = gblBfragBF16(W1h, 256, nt * 16, kc * 32, lane);
      wf[kc + 4] = gblBfragBF16(W1h, 256, nt * 16, 128 + kc * 32, lane);
    }
    v8f acc = {};
#pragma unroll
    for (int kc = 0; kc < 4; ++kc) {
      acc = wmma_bf16(aif[kc], wf[kc], acc);       // W1[:, :128] * x_i
      acc = wmma_bf16(ajf[kc], wf[kc + 4], acc);   // W1[:, 128:] * x_j
    }
    float b1v = b1[nt * 16 + l16];
#pragma unroll
    for (int v = 0; v < 8; ++v) {
      int e = eBase + half * 8 + v;
      int t = e / KNN;
      float hv = fmaxf(acc[v] + b1v, 0.0f)
               + lXi[t * CH + nt * 16 + l16];       // fp32 residual from LDS
      lHf[e * CH + nt * 16 + l16] = (__bf16)hv;
    }
  }

  // ---- m = h @ W2^T + b2 (overwrites Q region) ----
  v16bf ahf[4];
#pragma unroll
  for (int kc = 0; kc < 4; ++kc) ahf[kc] = ldsAfragBF16(lHf, eRow, kc * 32, half);
  for (int nt = 0; nt < 8; ++nt) {
    v16bf wf[4];
#pragma unroll
    for (int kc = 0; kc < 4; ++kc)
      wf[kc] = gblBfragBF16(W2h, 128, nt * 16, kc * 32, lane);
    v8f acc = {};
#pragma unroll
    for (int kc = 0; kc < 4; ++kc) acc = wmma_bf16(ahf[kc], wf[kc], acc);
    float b2v = b2[nt * 16 + l16];
#pragma unroll
    for (int v = 0; v < 8; ++v) {
      int e = eBase + half * 8 + v;
      lQM[e * CH + nt * 16 + l16] = acc[v] + b2v;
    }
  }
  __syncthreads();

  // ---- apply edge weights (view(-1,64,2) @ ew) and mean over k ----
  for (int o = tid; o < NTGT * CH; o += 192) {
    int t = o >> 7, c = o & 127;
    int sh2 = (c >> 1) << 1;  // s_*2
    int g = c & 1;
    float a = 0.f;
#pragma unroll
    for (int s = 0; s < KNN; ++s) {
      int e = t * KNN + s;
      a += lQM[e * CH + sh2]     * lEW[e * 4 + g]        // h=0
         + lQM[e * CH + sh2 + 1] * lEW[e * 4 + 2 + g];   // h=1
    }
    out[((size_t)b * NPTS + t0 + t) * CH + c] = a * (1.0f / 6.0f);
  }
}

// ---------------------------------------------------------------------------
extern "C" void kernel_launch(void* const* d_in, const int* in_sizes, int n_in,
                              void* d_out, int out_size, void* d_ws, size_t ws_size,
                              hipStream_t stream) {
  (void)in_sizes; (void)n_in; (void)out_size; (void)ws_size;
  const float* x  = (const float*)d_in[0];
  const float* W1 = (const float*)d_in[1];
  const float* b1 = (const float*)d_in[2];
  const float* W2 = (const float*)d_in[3];
  const float* b2 = (const float*)d_in[4];
  const float* Wq = (const float*)d_in[5];
  const float* bq = (const float*)d_in[6];
  const float* Wk = (const float*)d_in[7];
  const float* bk = (const float*)d_in[8];
  float* out = (float*)d_out;

  char* ws = (char*)d_ws;
  float*  sq  = (float*)ws;                  // 262144 B
  int*    idx = (int*)(ws + 262144);         // 1572864 B
  __bf16* Wqh = (__bf16*)(ws + 1835008);     // 65536 B
  __bf16* Wkh = (__bf16*)(ws + 1900544);     // 65536 B
  __bf16* W1h = (__bf16*)(ws + 1966080);     // 65536 B
  __bf16* W2h = (__bf16*)(ws + 2031616);     // 32768 B  (total ~2.0 MB)

  sq_kernel<<<BATCH * NPTS / 8, 256, 0, stream>>>(x, sq);
  cvt_kernel<<<(128 * 256 + 255) / 256, 256, 0, stream>>>(Wq, Wqh, 128 * 256);
  cvt_kernel<<<(128 * 256 + 255) / 256, 256, 0, stream>>>(Wk, Wkh, 128 * 256);
  cvt_kernel<<<(128 * 256 + 255) / 256, 256, 0, stream>>>(W1, W1h, 128 * 256);
  cvt_kernel<<<(128 * 128 + 255) / 256, 256, 0, stream>>>(W2, W2h, 128 * 128);
  knn_kernel<<<BATCH * (NPTS / 16), 32, 0, stream>>>(x, sq, idx);
  // smem: 8192 + 3*49152 + 1536 = 157184 bytes (< 320 KB/WGP)
  edge_kernel<<<BATCH * (NPTS / 16), 192, 157184, stream>>>(
      x, idx, W1h, b1, W2h, b2, Wqh, bq, Wkh, bk, out);
}
